// GABVNet_10763188044019
// MI455X (gfx1250) — compile-verified, matrix operand order checked by reading, not run on hardware
//
#include <hip/hip_runtime.h>
#include <math.h>

#define BB 64
#define NN 2048
#define TOT (BB*NN)

typedef __attribute__((ext_vector_type(16))) _Float16 v16h;
typedef __attribute__((ext_vector_type(8)))  _Float16 v8h;
typedef __attribute__((ext_vector_type(8)))  float    v8f;

#if defined(__has_builtin)
#if __has_builtin(__builtin_amdgcn_global_load_async_to_lds_b32)
#define HAVE_ASYNC 1
#endif
#endif
#ifndef HAVE_ASYNC
#define HAVE_ASYNC 0
#endif

typedef __attribute__((address_space(1))) int gas_i32;
typedef __attribute__((address_space(3))) int las_i32;

__device__ inline void wait_async0(){
#if defined(__has_builtin)
#if __has_builtin(__builtin_amdgcn_s_wait_asynccnt)
  __builtin_amdgcn_s_wait_asynccnt(0);
#else
  asm volatile("s_wait_asynccnt 0x0" ::: "memory");
#endif
#else
  asm volatile("s_wait_asynccnt 0x0" ::: "memory");
#endif
}

__device__ inline float sigm(float x){ return 1.f/(1.f+__expf(-x)); }
__device__ inline float softplusf(float x){ return x>20.f ? x : log1pf(expf(x)); }

// D = A(16x32 f16) x B(32x16 f16) + C(16x16 f32)
__device__ inline v8f wmma16(v16h a, v16h b, v8f c){
  return __builtin_amdgcn_wmma_f32_16x16x32_f16(false, a, false, b, (short)0, c, false, false);
}

// One 16x32 fragment row per lane, CDNA5 16-bit A/B layout:
// lanes 0-15: K 0..7 (V0-3) and 16..23 (V4-7); lanes 16-31: K 8..15 and 24..31.
__device__ inline v16h load_frag(const _Float16* rowp, int kbase, int lane){
  int kh = (lane & 16) ? 8 : 0;
  v8h lo = *(const v8h*)(rowp + kbase + kh);
  v8h hi = *(const v8h*)(rowp + kbase + 16 + kh);
  v16h r;
#pragma unroll
  for (int i = 0; i < 8; ++i){ r[i] = lo[i]; r[8+i] = hi[i]; }
  return r;
}

__device__ inline float block_reduce256(float v, float* red){
  int tid = threadIdx.x;
  red[tid] = v; __syncthreads();
#pragma unroll
  for (int s = 128; s > 0; s >>= 1){
    if (tid < s) red[tid] += red[tid + s];
    __syncthreads();
  }
  float r = red[0]; __syncthreads();
  return r;
}

// ---------------- small prep: reg MLP + nav MLP ----------------
__global__ void __launch_bounds__(64)
k_prep(const float* gamma, const float* rw1, const float* rb1, const float* rw2, const float* rb2,
       const float* nw1, const float* nb1, const float* lng, const float* lnb,
       const float* nw2, const float* nb2, const float* nw3, const float* nb3,
       const float* meta, const float* theta_in,
       float* lam, float* gpn, float* gnl, float* gth, float* theta_cur)
{
  int b = threadIdx.x;
  if (b >= BB) return;
  float g = gamma[b];
  float h16[16];
#pragma unroll
  for (int j = 0; j < 16; ++j){ float v = rw1[j]*g + rb1[j]; h16[j] = v > 0.f ? v : 0.f; }
  float o = rb2[0];
#pragma unroll
  for (int j = 0; j < 16; ++j) o += rw2[j]*h16[j];
  lam[b] = softplusf(o) + 1e-6f;

  float h1[64];
  for (int j = 0; j < 64; ++j){
    float v = nb1[j];
#pragma unroll
    for (int k = 0; k < 6; ++k) v += nw1[j*6+k]*meta[b*6+k];
    h1[j] = v;
  }
  float mu = 0.f; for (int j = 0; j < 64; ++j) mu += h1[j]; mu *= (1.f/64.f);
  float var = 0.f; for (int j = 0; j < 64; ++j){ float d = h1[j]-mu; var += d*d; } var *= (1.f/64.f);
  float inv = rsqrtf(var + 1e-5f);
  for (int j = 0; j < 64; ++j){ float v = (h1[j]-mu)*inv*lng[j] + lnb[j]; h1[j] = v > 0.f ? v : 0.f; }
  float h2[32];
  for (int j = 0; j < 32; ++j){
    float v = nb2[j];
    for (int k = 0; k < 64; ++k) v += nw2[j*64+k]*h1[k];
    h2[j] = v > 0.f ? v : 0.f;
  }
  float lg[4];
  for (int j = 0; j < 4; ++j){
    float v = nb3[j];
    for (int k = 0; k < 32; ++k) v += nw3[j*32+k]*h2[k];
    lg[j] = v;
  }
  gpn[b] = sigm(lg[0]);
  gnl[b] = softplusf(lg[1]) + 0.5f;
  gth[b] = sigm(lg[3]);
#pragma unroll
  for (int k = 0; k < 3; ++k) theta_cur[b*3+k] = theta_in[b*3+k];
}

// ---------------- channel diag (f64 phase, like reference) ----------------
__global__ void __launch_bounds__(256)
k_channel(const float* theta, const float* gain_, const float* yr, const float* yi,
          float* her, float* hei, float* xr, float* xi, int init)
{
  int i = blockIdx.x*256 + threadIdx.x;
  if (i >= TOT) return;
  int b = i / NN, t = i % NN;
  const double FSd = 1.0e10, FCd = 3.0e11, Cd = 3.0e8;
  const double PI2 = 6.283185307179586476925286766559;
  double R = (double)theta[b*3+0], V = (double)theta[b*3+1], A = (double)theta[b*3+2];
  double td = (double)t / FSd;
  double phase = PI2 * FCd * (R/Cd + (V/Cd)*td + 0.5*(A/Cd)*td*td);
  float cp = (float)cos(phase), sp = (float)sin(phase);
  float g = gain_[0];
  float Hr = g*cp, Hi = -g*sp;           // exp(-i*phase)*gain
  her[i] = Hr; hei[i] = Hi;
  if (init){
    float Yr = yr[i], Yi = yi[i];
    xr[i] = Yr*Hr + Yi*Hi;               // y * conj(h)
    xi[i] = Yi*Hr - Yr*Hi;
  }
}

// ---------------- global mean(|x|) ----------------
__global__ void k_zero1(float* a){ a[0] = 0.f; }

__global__ void __launch_bounds__(256)
k_absmean(const float* xr, const float* xi, float* acc)
{
  __shared__ float red[256];
  float s = 0.f;
  for (int j = blockIdx.x*256 + threadIdx.x; j < TOT; j += gridDim.x*256)
    s += sqrtf(xr[j]*xr[j] + xi[j]*xi[j]);
  s = block_reduce256(s, red);
  if (threadIdx.x == 0) atomicAdd(acc, s);
}

// ---------------- feats for PN tracker ----------------
__global__ void __launch_bounds__(256)
k_feats(const float* xr, const float* xi, const float* yr, const float* yi,
        const float* sumabs, float* fr, float* fi)
{
  int i = blockIdx.x*256 + threadIdx.x;
  if (i >= TOT) return;
  float mean = sumabs[0] * (1.f/(float)TOT);
  float Yr = yr[i], Yi = yi[i];
  if (mean > 0.1f){
    float Xr = xr[i], Xi = xi[i];
    float m = sqrtf(Xr*Xr + Xi*Xi) + 1e-6f;
    float nr = Xr/m, ni = Xi/m;
    fr[i] = Yr*nr + Yi*ni;               // y * conj(x_norm)
    fi[i] = Yi*nr - Yr*ni;
  } else {
    fr[i] = Yr; fi[i] = Yi;
  }
}

// ---------------- bidirectional GRU scan, v2 ----------------
// grid = 8: blockIdx = dir*4 + batch_group (16 batches each). block = 128 (4 waves).
// Wave w handles hidden slice hs=w (cols hs*16..hs*16+15) for ALL THREE gates:
// 6 persistent W_hh^T B-fragments in VGPRs, gate math on WMMA accumulators in
// registers, h_prev register-resident, ping-pong sH -> one barrier per step.
__global__ void __launch_bounds__(128)
k_gru(const float* fr, const float* fi,
      const float* whh_f, const float* wih_f, const float* bih_f, const float* bhh_f,
      const float* whh_b, const float* wih_b, const float* bih_b, const float* bhh_b,
      _Float16* out_f, _Float16* out_b)
{
  __shared__ _Float16 sW[192*64];        // W_hh f16 (gate-major): 24 KB
  __shared__ _Float16 sH[2][16*64];      // ping-pong hidden state: 2 x 2 KB
  __shared__ float    sF[2][2][16];      // ping-pong feats for current t

  const int dir = blockIdx.x >> 2;
  const int b0  = (blockIdx.x & 3) * 16;
  const float* whh = dir ? whh_b : whh_f;
  const float* wih = dir ? wih_b : wih_f;
  const float* bih = dir ? bih_b : bih_f;
  const float* bhh = dir ? bhh_b : bhh_f;
  _Float16* out = dir ? out_b : out_f;

  const int tid  = threadIdx.x;
  const int lane = tid & 31;
  const int hs   = tid >> 5;             // wave id = hidden slice
  const int ln   = lane & 15;

  for (int i = tid; i < 192*64; i += 128) sW[i] = (_Float16)whh[i];
  for (int i = tid; i < 16*64;  i += 128) sH[0][i] = (_Float16)0.f;
  {
    const int t0 = dir ? (NN-1) : 0;
    if (tid < 32){
      const int bl = tid & 15, wh = tid >> 4;
      const float* src = (wh ? fi : fr) + (size_t)(b0 + bl)*NN + t0;
      sF[0][wh][bl] = *src;
    }
  }
  __syncthreads();

  // persistent B fragments: rows (gate_ofs + hs*16 + ln) of W_hh, K chunks 0/32
  const int gr = hs*16 + ln, gz = 64 + gr, gn = 128 + gr;
  v16h bR0 = load_frag(sW + gr*64, 0, lane),  bR1 = load_frag(sW + gr*64, 32, lane);
  v16h bZ0 = load_frag(sW + gz*64, 0, lane),  bZ1 = load_frag(sW + gz*64, 32, lane);
  v16h bN0 = load_frag(sW + gn*64, 0, lane),  bN1 = load_frag(sW + gn*64, 32, lane);

  // per-lane gate weights (column = hs*16+ln fixed per lane)
  const float wr0 = wih[2*gr], wr1 = wih[2*gr+1], bri = bih[gr], brh = bhh[gr];
  const float wz0 = wih[2*gz], wz1 = wih[2*gz+1], bzi = bih[gz], bzh = bhh[gz];
  const float wn0 = wih[2*gn], wn1 = wih[2*gn+1], bni = bih[gn], bnh = bhh[gn];

  float hreg[8];
#pragma unroll
  for (int j = 0; j < 8; ++j) hreg[j] = 0.f;

  const int hcol = hs*16 + ln;           // this lane's hidden column

  auto do_step = [&](int bc, int bn, int t){
    // recurrent matmul on previous hidden state
    v16h a0 = load_frag(&sH[bc][ln*64], 0,  lane);
    v16h a1 = load_frag(&sH[bc][ln*64], 32, lane);
    v8f aR = {}, aZ = {}, aN = {};
    aR = wmma16(a0, bR0, aR);  aR = wmma16(a1, bR1, aR);
    aZ = wmma16(a0, bZ0, aZ);  aZ = wmma16(a1, bZ1, aZ);
    aN = wmma16(a0, bN0, aN);  aN = wmma16(a1, bN1, aN);

    // stage next step's feats into the other buffer (async direct-to-LDS)
    const int tn = dir ? (t-1) : (t+1);
    if (tid < 32 && (unsigned)tn < NN){
      const int bl = tid & 15, wh = tid >> 4;
      const float* src = (wh ? fi : fr) + (size_t)(b0 + bl)*NN + tn;
#if HAVE_ASYNC
      __builtin_amdgcn_global_load_async_to_lds_b32(
          (gas_i32*)src, (las_i32*)&sF[bn][wh][bl], 0, 0);
#else
      sF[bn][wh][bl] = *src;
#endif
      const int tp = dir ? (t-2) : (t+2);
      if ((unsigned)tp < NN)
        __builtin_prefetch((wh ? fi : fr) + (size_t)(b0 + bl)*NN + tp, 0, 0);
    }

    // gate nonlinearity on accumulators (rows = D-tile layout positions)
    const int rb = (lane >> 4) << 3;
#pragma unroll
    for (int j = 0; j < 8; ++j){
      const int row = j + rb;                       // local batch row 0..15
      const float f0 = sF[bc][0][row], f1 = sF[bc][1][row];
      const float xr = f0*wr0 + f1*wr1 + bri;
      const float xz = f0*wz0 + f1*wz1 + bzi;
      const float xn = f0*wn0 + f1*wn1 + bni;
      const float r  = sigm(xr + aR[j] + brh);
      const float z  = sigm(xz + aZ[j] + bzh);
      const float nn = tanhf(xn + r*(aN[j] + bnh));
      const float hnew = (1.f - z)*nn + z*hreg[j];
      hreg[j] = hnew;
      sH[bn][row*64 + hcol] = (_Float16)hnew;
      out[((size_t)(b0 + row)*NN + t)*64 + hcol] = (_Float16)hnew;
    }
    wait_async0();
    __syncthreads();
  };

  for (int s = 0; s < NN; s += 2){
    const int t0 = dir ? (NN-1-s) : s;
    const int t1 = dir ? (t0-1) : (t0+1);
    do_step(0, 1, t0);
    do_step(1, 0, t1);
  }
}

// ---------------- head MLP + derot + y_eff  (WMMA GEMM over 131072 rows) ----------------
__global__ void __launch_bounds__(256)
k_head(const _Float16* out_f, const _Float16* out_b,
       const float* w1, const float* b1, const float* w2, const float* b2,
       const float* gpn, const float* phase_bias,
       const float* yr, const float* yi, float* yer, float* yei)
{
  __shared__ _Float16 sW1[32*128];       // 8 KB
  __shared__ float sB1[32], sW2[64], sB2[2];
  __shared__ float hbuf[8][16][32];      // 16 KB

  const int tid = threadIdx.x;
  for (int i = tid; i < 32*128; i += 256) sW1[i] = (_Float16)w1[i];
  if (tid < 32) sB1[tid] = b1[tid];
  if (tid < 64) sW2[tid] = w2[tid];
  if (tid < 2)  sB2[tid] = b2[tid];
  __syncthreads();

  const int lane = tid & 31, wave = tid >> 5;
  const size_t tile = (size_t)blockIdx.x*8 + wave;   // 0..8191
  const size_t row0 = tile*16;

  v8f acc0 = {}, acc1 = {};
#pragma unroll
  for (int c = 0; c < 4; ++c){
    const int kb = c*32;
    const _Float16* src = (kb < 64) ? out_f : out_b;
    const int kk = (kb < 64) ? kb : kb - 64;
    v16h a  = load_frag(src + (row0 + (lane & 15))*64, kk, lane);
    v16h w0 = load_frag(sW1 + (0  + (lane & 15))*128, kb, lane);
    v16h w1f= load_frag(sW1 + (16 + (lane & 15))*128, kb, lane);
    acc0 = wmma16(a, w0,  acc0);
    acc1 = wmma16(a, w1f, acc1);
  }

#pragma unroll
  for (int j = 0; j < 8; ++j){
    const int r  = j + ((lane >> 4) << 3);
    const int c0 = lane & 15;
    float v0 = acc0[j] + sB1[c0];
    float v1 = acc1[j] + sB1[16 + c0];
    hbuf[wave][r][c0]      = v0 > 0.f ? v0 : 0.f;
    hbuf[wave][r][16 + c0] = v1 > 0.f ? v1 : 0.f;
  }
  __syncthreads();

  if (lane < 16){
    const size_t row = row0 + lane;
    const int b = (int)(row >> 11);
    float c0 = sB2[0], c1 = sB2[1];
#pragma unroll
    for (int k = 0; k < 32; ++k){
      const float h = hbuf[wave][lane][k];
      c0 += h*sW2[k];
      c1 += h*sW2[32 + k];
    }
    const float mag = sqrtf(c0*c0 + c1*c1 + 1e-8f);
    const float cphi = c0/mag, sphi = c1/mag;
    const float pb = phase_bias[0];
    const float cb = cosf(pb), sb = sinf(pb);
    // derot = (cphi - i*sphi) * exp(-i*pb)
    const float dr = cphi*cb - sphi*sb;
    const float di = -cphi*sb - sphi*cb;
    const float g  = gpn[b];
    float er = (1.f - g) + g*dr;
    float ei = g*di;
    const float em = sqrtf(er*er + ei*ei) + 1e-8f;
    er /= em; ei /= em;
    const float Yr = yr[row], Yi = yi[row];
    yer[row] = Yr*er - Yi*ei;
    yei[row] = Yr*ei + Yi*er;
  }
}

// ---------------- CG solve (diagonal A) + tanh + power stats ----------------
__global__ void __launch_bounds__(256)
k_cg(const float* her, const float* hei, const float* yer, const float* yei,
     const float* lam_, const float* gnl, const float* scale_slope,
     float* xr_, float* xi_, float* rp_, float* xq_)
{
  __shared__ float red[256];
  const int b = blockIdx.x, tid = threadIdx.x;
  const float lam = lam_[b];
  float hd[8], br[8], bi[8], xr[8], xi[8], rr[8], ri[8], pr[8], pi[8];
  float rs = 0.f;
#pragma unroll
  for (int k = 0; k < 8; ++k){
    const size_t idx = (size_t)b*NN + tid + k*256;
    const float Hr = her[idx], Hi = hei[idx];
    const float Yr = yer[idx], Yi = yei[idx];
    hd[k] = Hr*Hr + Hi*Hi;
    br[k] = Yr*Hr + Yi*Hi;
    bi[k] = Yi*Hr - Yr*Hi;
    xr[k] = 0.f; xi[k] = 0.f;
    rr[k] = br[k]; ri[k] = bi[k];
    pr[k] = br[k]; pi[k] = bi[k];
    rs += br[k]*br[k] + bi[k]*bi[k];
  }
  rs = block_reduce256(rs, red);
  for (int it = 0; it < 5; ++it){
    float pap = 0.f;
#pragma unroll
    for (int k = 0; k < 8; ++k){
      const float d = hd[k] + lam;
      pap += d*(pr[k]*pr[k] + pi[k]*pi[k]);
    }
    pap = block_reduce256(pap, red);
    const float alpha = rs / (pap + 1e-12f);
    float rsn = 0.f;
#pragma unroll
    for (int k = 0; k < 8; ++k){
      const float d = hd[k] + lam;
      xr[k] += alpha*pr[k];  xi[k] += alpha*pi[k];
      rr[k] -= alpha*d*pr[k]; ri[k] -= alpha*d*pi[k];
      rsn += rr[k]*rr[k] + ri[k]*ri[k];
    }
    rsn = block_reduce256(rsn, red);
    const float beta = rsn / (rs + 1e-12f);
#pragma unroll
    for (int k = 0; k < 8; ++k){ pr[k] = rr[k] + beta*pr[k]; pi[k] = ri[k] + beta*pi[k]; }
    rs = rsn;
  }
  const float bnl = scale_slope[0]*gnl[b];
  float sres = 0.f, sx = 0.f;
#pragma unroll
  for (int k = 0; k < 8; ++k){
    const size_t idx = (size_t)b*NN + tid + k*256;
    const float Xr = tanhf(bnl*xr[k]), Xi = tanhf(bnl*xi[k]);
    xr_[idx] = Xr; xi_[idx] = Xi;
    const float Hr = her[idx], Hi = hei[idx];
    const float rre = yer[idx] - (Xr*Hr - Xi*Hi);
    const float rim = yei[idx] - (Xr*Hi + Xi*Hr);
    sres += rre*rre + rim*rim;
    sx   += Xr*Xr + Xi*Xi;
  }
  sres = block_reduce256(sres, red);
  sx   = block_reduce256(sx, red);
  if (tid == 0){
    rp_[b] = log1pf(sres * (1.f/NN));
    xq_[b] = log1pf(sx   * (1.f/NN));
  }
}

// ---------------- theta update MLP ----------------
__global__ void __launch_bounds__(64)
k_theta(const float* tw1, const float* tb1, const float* tw2, const float* tb2,
        const float* tw3, const float* tb3, const float* lr, const float* gth,
        const float* rp, const float* xq, const float* gamma, float* theta)
{
  const int b = threadIdx.x;
  if (b >= BB) return;
  const float thmin[3] = {4.9e8f, -1.0e4f, -1.0e2f};
  const float thmax[3] = {5.1e8f,  1.0e4f,  1.0e2f};
  const float maxd[3]  = {20.f, 5.f, 1.f};
  float f[6];
#pragma unroll
  for (int k = 0; k < 3; ++k) f[k] = (theta[b*3+k] - thmin[k])/(thmax[k] - thmin[k]);
  f[3] = rp[b]; f[4] = xq[b]; f[5] = gamma[b];
  float g1[32];
  for (int j = 0; j < 32; ++j){
    float v = tb1[j];
#pragma unroll
    for (int k = 0; k < 6; ++k) v += tw1[j*6+k]*f[k];
    g1[j] = v > 0.f ? v : 0.f;
  }
  float g2[32];
  for (int j = 0; j < 32; ++j){
    float v = tb2[j];
    for (int k = 0; k < 32; ++k) v += tw2[j*32+k]*g1[k];
    g2[j] = v > 0.f ? v : 0.f;
  }
#pragma unroll
  for (int j = 0; j < 3; ++j){
    float v = tb3[j];
    for (int k = 0; k < 32; ++k) v += tw3[j*32+k]*g2[k];
    const float d = tanhf(v)*maxd[j]*lr[0]*gth[b];
    float nt = theta[b*3+j] + d;
    nt = fminf(fmaxf(nt, thmin[j]), thmax[j]);
    theta[b*3+j] = nt;
  }
}

// ---------------- pack output ----------------
__global__ void __launch_bounds__(256)
k_out(const float* xr, const float* xi, float* out)
{
  int i = blockIdx.x*256 + threadIdx.x;
  if (i >= TOT) return;
  out[2*i]   = xr[i];
  out[2*i+1] = xi[i];
}

extern "C" void kernel_launch(void* const* d_in, const int* in_sizes, int n_in,
                              void* d_out, int out_size, void* d_ws, size_t ws_size,
                              hipStream_t stream)
{
  (void)in_sizes; (void)n_in; (void)out_size; (void)ws_size;
  // input order = setup_inputs dict order (params flattened in insertion order)
  const float* yqr   = (const float*)d_in[0];
  const float* yqi   = (const float*)d_in[1];
  const float* theta = (const float*)d_in[2];
  const float* meta  = (const float*)d_in[3];
  const float* gamma = (const float*)d_in[4];
  const float* scale_slope = (const float*)d_in[5];
  const float* buss  = (const float*)d_in[6];
  const float* rw1 = (const float*)d_in[7],  *rb1 = (const float*)d_in[8];
  const float* rw2 = (const float*)d_in[9],  *rb2 = (const float*)d_in[10];
  const float* nw1 = (const float*)d_in[11], *nb1 = (const float*)d_in[12];
  const float* lng = (const float*)d_in[13], *lnb = (const float*)d_in[14];
  const float* nw2 = (const float*)d_in[15], *nb2 = (const float*)d_in[16];
  const float* nw3 = (const float*)d_in[17], *nb3 = (const float*)d_in[18];
  const float* wih_f = (const float*)d_in[19], *whh_f = (const float*)d_in[20];
  const float* bih_f = (const float*)d_in[21], *bhh_f = (const float*)d_in[22];
  const float* wih_b = (const float*)d_in[23], *whh_b = (const float*)d_in[24];
  const float* bih_b = (const float*)d_in[25], *bhh_b = (const float*)d_in[26];
  const float* hw1 = (const float*)d_in[27], *hb1 = (const float*)d_in[28];
  const float* hw2 = (const float*)d_in[29], *hb2 = (const float*)d_in[30];
  const float* pbias = (const float*)d_in[31];
  const float* tw1 = (const float*)d_in[32], *tb1 = (const float*)d_in[33];
  const float* tw2 = (const float*)d_in[34], *tb2 = (const float*)d_in[35];
  const float* tw3 = (const float*)d_in[36], *tb3 = (const float*)d_in[37];
  const float* lr  = (const float*)d_in[38];

  float* W = (float*)d_ws;
  float* XR  = W;             float* XI  = W + 1*TOT;
  float* HER = W + 2*TOT;     float* HEI = W + 3*TOT;
  float* YER = W + 4*TOT;     float* YEI = W + 5*TOT;
  float* FR  = W + 6*TOT;     float* FI  = W + 7*TOT;
  float* SM  = W + 8*TOT;     // small block
  float* LAM = SM;       float* GPN = SM + 64;  float* GNL = SM + 128;
  float* GTH = SM + 192; float* RP  = SM + 256; float* XQ  = SM + 320;
  float* THC = SM + 384; float* ACC = SM + 576;
  _Float16* OUTF = (_Float16*)(SM + 1024);
  _Float16* OUTB = OUTF + (size_t)TOT*64;

  const int EB = (TOT + 255)/256;   // 512 elementwise blocks

  k_prep<<<1, 64, 0, stream>>>(gamma, rw1, rb1, rw2, rb2, nw1, nb1, lng, lnb,
                               nw2, nb2, nw3, nb3, meta, theta,
                               LAM, GPN, GNL, GTH, THC);
  k_channel<<<EB, 256, 0, stream>>>(THC, buss, yqr, yqi, HER, HEI, XR, XI, 1);

  for (int layer = 0; layer < 4; ++layer){
    k_zero1<<<1, 1, 0, stream>>>(ACC);
    k_absmean<<<64, 256, 0, stream>>>(XR, XI, ACC);
    k_feats<<<EB, 256, 0, stream>>>(XR, XI, yqr, yqi, ACC, FR, FI);
    k_gru<<<8, 128, 0, stream>>>(FR, FI, whh_f, wih_f, bih_f, bhh_f,
                                 whh_b, wih_b, bih_b, bhh_b, OUTF, OUTB);
    k_head<<<1024, 256, 0, stream>>>(OUTF, OUTB, hw1, hb1, hw2, hb2,
                                     GPN, pbias, yqr, yqi, YER, YEI);
    k_cg<<<64, 256, 0, stream>>>(HER, HEI, YER, YEI, LAM, GNL, scale_slope,
                                 XR, XI, RP, XQ);
    if (layer >= 1){
      k_theta<<<1, 64, 0, stream>>>(tw1, tb1, tw2, tb2, tw3, tb3, lr, GTH,
                                    RP, XQ, gamma, THC);
      k_channel<<<EB, 256, 0, stream>>>(THC, buss, yqr, yqi, HER, HEI, XR, XI, 0);
    }
  }
  k_out<<<EB, 256, 0, stream>>>(XR, XI, (float*)d_out);
}
